// GroupedQueryAttention_34892314312808
// MI455X (gfx1250) — matrix operand order repaired and checked by reading, not compiled, not run on hardware
//
#include <hip/hip_runtime.h>
#include <hip/hip_bf16.h>

// ---------------------------------------------------------------------------
// GQA attention layer for MI455X (gfx1250, wave32, WMMA).
// Pipeline: f32->bf16 convert -> QKV GEMMs (wmma bf16, double-buffered)
// -> flash attention (wmma bf16, DPP-reduced online softmax, split
// unmasked/masked kv loop) -> output GEMM (f32 out).
// ---------------------------------------------------------------------------

#define B_  2
#define S_  2048
#define E_  1024
#define H_  16
#define HKV_ 4
#define D_  64
#define REP_ 4
#define KVE_ (HKV_ * D_)       // 256
#define M_  (B_ * S_)          // 4096 rows

typedef __attribute__((ext_vector_type(16))) __bf16 bf16x16;
typedef __attribute__((ext_vector_type(8)))  __bf16 bf16x8;
typedef __attribute__((ext_vector_type(8)))  float  f32x8;

__device__ __forceinline__ f32x8 zero8() {
  f32x8 z = {0.f, 0.f, 0.f, 0.f, 0.f, 0.f, 0.f, 0.f};
  return z;
}

__device__ __forceinline__ f32x8 wmma_bf16(bf16x16 a, bf16x16 b, f32x8 c) {
  // D = A(16x32) * B(32x16) + C, fp32 accumulate
  return __builtin_amdgcn_wmma_f32_16x16x32_bf16(
      false, a, false, b, (short)0, c, false, false);
}

// A-fragment (16x32 bf16): lane (half, mr) holds row mr;
// elems 0..15 = K[k0 + half*16 .. +15] contiguous.
__device__ __forceinline__ bf16x16 load_a_frag(const __bf16* row, int k0, int half) {
  union { bf16x16 v; bf16x8 h[2]; } u;
  u.h[0] = *(const bf16x8*)(row + k0 + half * 16);
  u.h[1] = *(const bf16x8*)(row + k0 + half * 16 + 8);
  return u.v;
}

// B-fragment (32x16 bf16) from a row-major [N][K] matrix (i.e. B = W^T):
// lane (half, mr) holds column mr, K[k0 + half*16 .. +15] contiguous.
__device__ __forceinline__ bf16x16 load_b_frag(const __bf16* row, int k0, int half) {
  return *(const bf16x16*)(row + k0 + half * 16);
}

// ---------------------------------------------------------------------------
// DPP16 butterfly reduction within each 16-lane half (matches WMMA C layout).
// Pure VALU: no LDS traffic, no dscnt waits.
// ---------------------------------------------------------------------------
template <int CTRL>
__device__ __forceinline__ float dpp_movf(float x) {
  return __int_as_float(__builtin_amdgcn_update_dpp(
      0, __float_as_int(x), CTRL, 0xF, 0xF, true));
}
__device__ __forceinline__ float rowmax16(float x) {
  x = fmaxf(x, dpp_movf<0xB1>(x));    // quad_perm(1,0,3,2)
  x = fmaxf(x, dpp_movf<0x4E>(x));    // quad_perm(2,3,0,1)
  x = fmaxf(x, dpp_movf<0x141>(x));   // row_half_mirror
  x = fmaxf(x, dpp_movf<0x140>(x));   // row_mirror
  return x;
}
__device__ __forceinline__ float rowsum16(float x) {
  x += dpp_movf<0xB1>(x);
  x += dpp_movf<0x4E>(x);
  x += dpp_movf<0x141>(x);
  x += dpp_movf<0x140>(x);
  return x;
}

// Raw v_exp_f32: args in (-inf, 0]; flush-to-zero is the desired behavior.
__device__ __forceinline__ float exp2_raw(float x) {
  return __builtin_amdgcn_exp2f(x);
}

// ---------------------------------------------------------------------------
// fp32 -> bf16 conversion
// ---------------------------------------------------------------------------
__global__ void k_f32_to_bf16(const float* __restrict__ in,
                              __bf16* __restrict__ out, int n) {
  int i = blockIdx.x * blockDim.x + threadIdx.x;
  if (i < n) out[i] = (__bf16)in[i];
}

// ---------------------------------------------------------------------------
// GEMM: C[M,N] = A[M,K](bf16) * W[N,K]^T(bf16) + bias[N](f32)
// Block: 256 threads = 8 waves in 2(M) x 4(N); wave tile 32x32 (2x2 WMMA).
// k-loop is software pipelined: next tile's fragments load under this tile's
// WMMAs. Grid: (N/128, M/64).
// ---------------------------------------------------------------------------
template <bool STORE_BF16>
__global__ void k_gemm_wmma(const __bf16* __restrict__ A,
                            const __bf16* __restrict__ W,
                            const float* __restrict__ bias,
                            void* __restrict__ C, int N, int K) {
  const int wave = threadIdx.x >> 5;
  const int lane = threadIdx.x & 31;
  const int half = lane >> 4;
  const int mr   = lane & 15;
  const int wm = wave >> 2;               // 0..1
  const int wn = wave & 3;                // 0..3
  const int m0 = blockIdx.y * 64 + wm * 32;
  const int n0 = blockIdx.x * 128 + wn * 32;

  const __bf16* arow0 = A + (size_t)(m0 + mr) * K;
  const __bf16* arow1 = A + (size_t)(m0 + 16 + mr) * K;
  const __bf16* wrow0 = W + (size_t)(n0 + mr) * K;
  const __bf16* wrow1 = W + (size_t)(n0 + 16 + mr) * K;

  f32x8 acc00 = zero8(), acc01 = zero8(), acc10 = zero8(), acc11 = zero8();

  // prologue: load first k tile
  bf16x16 a0 = load_a_frag(arow0, 0, half);
  bf16x16 a1 = load_a_frag(arow1, 0, half);
  bf16x16 b0 = load_b_frag(wrow0, 0, half);
  bf16x16 b1 = load_b_frag(wrow1, 0, half);

  for (int k0 = 32; k0 < K; k0 += 32) {
    bf16x16 na0 = load_a_frag(arow0, k0, half);
    bf16x16 na1 = load_a_frag(arow1, k0, half);
    bf16x16 nb0 = load_b_frag(wrow0, k0, half);
    bf16x16 nb1 = load_b_frag(wrow1, k0, half);
    acc00 = wmma_bf16(a0, b0, acc00);
    acc01 = wmma_bf16(a0, b1, acc01);
    acc10 = wmma_bf16(a1, b0, acc10);
    acc11 = wmma_bf16(a1, b1, acc11);
    a0 = na0; a1 = na1; b0 = nb0; b1 = nb1;
  }
  acc00 = wmma_bf16(a0, b0, acc00);
  acc01 = wmma_bf16(a0, b1, acc01);
  acc10 = wmma_bf16(a1, b0, acc10);
  acc11 = wmma_bf16(a1, b1, acc11);

  // C layout: lane half/mr -> col n0+ni*16+mr ; VGPR r -> row m0+mi*16+half*8+r
  #pragma unroll
  for (int mi = 0; mi < 2; ++mi) {
    #pragma unroll
    for (int ni = 0; ni < 2; ++ni) {
      const f32x8 acc = (mi == 0) ? (ni == 0 ? acc00 : acc01)
                                  : (ni == 0 ? acc10 : acc11);
      const int col = n0 + ni * 16 + mr;
      const float bv = bias[col];
      #pragma unroll
      for (int r = 0; r < 8; ++r) {
        const int row = m0 + mi * 16 + half * 8 + r;
        const float v = acc[r] + bv;
        if (STORE_BF16)
          ((__bf16*)C)[(size_t)row * N + col] = (__bf16)v;
        else
          ((float*)C)[(size_t)row * N + col] = v;
      }
    }
  }
}

// ---------------------------------------------------------------------------
// One kv tile (32 keys) of flash attention for one wave's 16-row Q tile.
// MASKED is compile-time: only the final (diagonal-crossing) tile pays for
// the causal compare/select; all other tiles run the lean path.
// kp -> K row (t0+mr), vp -> V row (t0+lane); both advance by 32*KVE outside.
// ---------------------------------------------------------------------------
template <bool MASKED>
__device__ __forceinline__ void kv_step(
    const __bf16* __restrict__ kp, const __bf16* __restrict__ vp,
    const bf16x16& aq0, const bf16x16& aq1,
    f32x8 (&acc)[4], float (&mrow)[8], float (&lrow)[8],
    int trel, int half, int mr, int lane,
    __bf16* __restrict__ ldsPw, __bf16* __restrict__ ldsVtw) {
  const float sc = 0.125f * 1.44269504088896f;   // D^-0.5 * log2(e)

  // ---- scores S = Q(16x64) @ K^T(64x32) ----
  bf16x16 bk00 = load_b_frag(kp, 0, half);
  bf16x16 bk01 = load_b_frag(kp, 32, half);
  bf16x16 bk10 = load_b_frag(kp + 16 * KVE_, 0, half);
  bf16x16 bk11 = load_b_frag(kp + 16 * KVE_, 32, half);
  f32x8 s0 = zero8(), s1 = zero8();
  s0 = wmma_bf16(aq0, bk00, s0);
  s0 = wmma_bf16(aq1, bk01, s0);
  s1 = wmma_bf16(aq0, bk10, s1);
  s1 = wmma_bf16(aq1, bk11, s1);

  // ---- stage V tile (32 kv x 64 d) transposed into LDS: Vt[d][kv] ----
  // Issued before the softmax so the global-load latency hides under it.
  #pragma unroll
  for (int c4 = 0; c4 < 4; ++c4) {
    bf16x16 vv = *(const bf16x16*)(vp + c4 * 16);
    #pragma unroll
    for (int e = 0; e < 16; ++e)
      ldsVtw[(c4 * 16 + e) * 32] = vv[e];          // base already + lane
  }

  // ---- online softmax (exp2 domain), per-row DPP reduce over 16 lanes ----
  float alpha[8];
  #pragma unroll
  for (int r = 0; r < 8; ++r) {
    const int rr = half * 8 + r;                    // row within 16-row tile
    float v0 = s0[r] * sc;
    float v1 = s1[r] * sc;
    if (MASKED) {
      if (trel + mr > rr)      v0 = -__builtin_inff();
      if (trel + 16 + mr > rr) v1 = -__builtin_inff();
    }
    const float mx = rowmax16(fmaxf(v0, v1));
    const float mn = fmaxf(mrow[r], mx);
    alpha[r] = exp2_raw(mrow[r] - mn);
    mrow[r] = mn;
    const float p0 = exp2_raw(v0 - mn);
    const float p1 = exp2_raw(v1 - mn);
    lrow[r] = lrow[r] * alpha[r] + rowsum16(p0 + p1);
    ldsPw[rr * 32] = (__bf16)p0;                    // base already + mr
    ldsPw[rr * 32 + 16] = (__bf16)p1;
  }
  #pragma unroll
  for (int n4 = 0; n4 < 4; ++n4)
    #pragma unroll
    for (int r = 0; r < 8; ++r)
      acc[n4][r] *= alpha[r];

  // ---- ctx += P(16x32) @ V(32x64); P re-read as A-fragment from LDS ----
  const bf16x16 ap = load_a_frag(ldsPw - mr + (size_t)mr * 32, 0, half);
  const __bf16* vt0 = ldsVtw - lane;                // wave base of Vt
  #pragma unroll
  for (int n4 = 0; n4 < 4; ++n4) {
    const bf16x16 bv = *(const bf16x16*)(vt0 + (n4 * 16 + mr) * 32 + half * 16);
    acc[n4] = wmma_bf16(ap, bv, acc[n4]);
  }
}

// ---------------------------------------------------------------------------
// Flash attention (causal, GQA). One wave owns a 16-row Q tile; kv tiles of 32.
// Grid: (S/128, B*H), block 256 threads (8 waves).
// ---------------------------------------------------------------------------
__global__ void k_gqa_flash(const __bf16* __restrict__ Qb,
                            const __bf16* __restrict__ Kb,
                            const __bf16* __restrict__ Vb,
                            __bf16* __restrict__ Ctx) {
  __shared__ __bf16 ldsP[8][16 * 32];     // P tile per wave (8 KB)
  __shared__ __bf16 ldsVt[8][64 * 32];    // V^T tile per wave (32 KB)

  const int wave = threadIdx.x >> 5;
  const int lane = threadIdx.x & 31;
  const int half = lane >> 4;
  const int mr   = lane & 15;

  const int bh = blockIdx.y;
  const int b  = bh / H_;
  const int h  = bh % H_;
  const int g  = h / REP_;                // kv head group
  const int qs = blockIdx.x * 128 + wave * 16;

  // Q A-fragments (fixed for whole kv loop): rows qs..qs+15, d = 0..63
  const __bf16* qrow = Qb + (size_t)(b * S_ + qs + mr) * E_ + h * D_;
  const bf16x16 aq0 = load_a_frag(qrow, 0, half);
  const bf16x16 aq1 = load_a_frag(qrow, 32, half);

  f32x8 acc[4];
  #pragma unroll
  for (int i = 0; i < 4; ++i) acc[i] = zero8();
  float mrow[8], lrow[8];
  #pragma unroll
  for (int r = 0; r < 8; ++r) { mrow[r] = -__builtin_inff(); lrow[r] = 0.f; }

  // incrementing row pointers (no per-iteration 64-bit address rebuild)
  const __bf16* kp = Kb + (size_t)(b * S_ + mr) * KVE_ + g * D_;
  const __bf16* vp = Vb + (size_t)(b * S_ + lane) * KVE_ + g * D_;
  __bf16* ldsPw  = &ldsP[wave][mr];
  __bf16* ldsVtw = &ldsVt[wave][lane];

  // last kv tile is the only one that can cross the causal diagonal
  const int t_last = ((qs + 15) >> 5) << 5;

  for (int t0 = 0; t0 < t_last; t0 += 32) {
    kv_step<false>(kp, vp, aq0, aq1, acc, mrow, lrow,
                   0, half, mr, lane, ldsPw, ldsVtw);
    kp += 32 * KVE_;
    vp += 32 * KVE_;
  }
  kv_step<true>(kp, vp, aq0, aq1, acc, mrow, lrow,
                t_last - qs, half, mr, lane, ldsPw, ldsVtw);

  // ---- epilogue: normalize (v_rcp) and store ctx (bf16) ----
  #pragma unroll
  for (int n4 = 0; n4 < 4; ++n4) {
    const int col = h * D_ + n4 * 16 + mr;
    #pragma unroll
    for (int r = 0; r < 8; ++r) {
      const int row = b * S_ + qs + half * 8 + r;
      const float v = acc[n4][r] * __builtin_amdgcn_rcpf(lrow[r]);
      Ctx[(size_t)row * E_ + col] = (__bf16)v;
    }
  }
}

// ---------------------------------------------------------------------------
// launch
// ---------------------------------------------------------------------------
extern "C" void kernel_launch(void* const* d_in, const int* in_sizes, int n_in,
                              void* d_out, int out_size, void* d_ws, size_t ws_size,
                              hipStream_t stream) {
  (void)in_sizes; (void)n_in; (void)out_size; (void)ws_size;
  const float* x    = (const float*)d_in[0];
  /* d_in[1] = mask: causal, computed analytically */
  const float* wq_w = (const float*)d_in[2];
  const float* wq_b = (const float*)d_in[3];
  const float* wk_w = (const float*)d_in[4];
  const float* wk_b = (const float*)d_in[5];
  const float* wv_w = (const float*)d_in[6];
  const float* wv_b = (const float*)d_in[7];
  const float* wo_w = (const float*)d_in[8];
  const float* wo_b = (const float*)d_in[9];
  float* out = (float*)d_out;

  // workspace carve-up (256B aligned)
  size_t off = 0;
  auto carve = [&](size_t elems) {
    void* p = (char*)d_ws + off;
    off += (elems * sizeof(__bf16) + 255) & ~(size_t)255;
    return (__bf16*)p;
  };
  __bf16* xbf  = carve((size_t)M_ * E_);
  __bf16* wqbf = carve((size_t)E_ * E_);
  __bf16* wkbf = carve((size_t)KVE_ * E_);
  __bf16* wvbf = carve((size_t)KVE_ * E_);
  __bf16* wobf = carve((size_t)E_ * E_);
  __bf16* qbf  = carve((size_t)M_ * E_);
  __bf16* kbf  = carve((size_t)M_ * KVE_);
  __bf16* vbf  = carve((size_t)M_ * KVE_);
  __bf16* cbf  = carve((size_t)M_ * E_);

  auto cvt = [&](const float* src, __bf16* dst, int n) {
    k_f32_to_bf16<<<(n + 255) / 256, 256, 0, stream>>>(src, dst, n);
  };
  cvt(x,    xbf,  M_ * E_);
  cvt(wq_w, wqbf, E_ * E_);
  cvt(wk_w, wkbf, KVE_ * E_);
  cvt(wv_w, wvbf, KVE_ * E_);
  cvt(wo_w, wobf, E_ * E_);

  // Q/K/V projections (bf16 out)
  k_gemm_wmma<true><<<dim3(E_ / 128,  M_ / 64), 256, 0, stream>>>(xbf, wqbf, wq_b, qbf, E_,  E_);
  k_gemm_wmma<true><<<dim3(KVE_ / 128, M_ / 64), 256, 0, stream>>>(xbf, wkbf, wk_b, kbf, KVE_, E_);
  k_gemm_wmma<true><<<dim3(KVE_ / 128, M_ / 64), 256, 0, stream>>>(xbf, wvbf, wv_b, vbf, KVE_, E_);

  // attention
  k_gqa_flash<<<dim3(S_ / 128, B_ * H_), 256, 0, stream>>>(qbf, kbf, vbf, cbf);

  // output projection (f32 out, straight to d_out)
  k_gemm_wmma<false><<<dim3(E_ / 128, M_ / 64), 256, 0, stream>>>(cbf, wobf, wo_b, out, E_, E_);
}
